// REMed_12816182411528
// MI455X (gfx1250) — matrix operand order repaired and checked by reading, not compiled, not run on hardware
//
#include <hip/hip_runtime.h>
#include <math.h>

// ---------------------------------------------------------------------------
// CDNA5 / gfx1250 WMMA types & helpers (wave32)
// ---------------------------------------------------------------------------
typedef __attribute__((ext_vector_type(16))) __bf16 v16bf;
typedef __attribute__((ext_vector_type(8)))  float  v8f;

#define WMMA_BF16(a, b, c) \
  __builtin_amdgcn_wmma_f32_16x16x32_bf16(false, (a), false, (b), (short)0, (c), false, false)

// A operand: M x K row-major bf16, 16x32 tile.
// lanes 0-15: M=lane, elems 0..15 -> K=k0+0..15
// lanes 16-31: M=lane-16, elems 0..15 -> K=k0+16..31
__device__ __forceinline__ v16bf frag_a(const __bf16* base, int ld, int m0, int k0, int lane) {
  const int half = (lane >> 4) & 1;
  const __bf16* p = base + (size_t)(m0 + (lane & 15)) * ld + k0 + half * 16;
  v16bf f;
#pragma unroll
  for (int i = 0; i < 8; ++i) f[i] = p[i];
#pragma unroll
  for (int i = 0; i < 8; ++i) f[i + 8] = p[8 + i];
  return f;
}

// B operand sourced from an N-major [N x Kdim] row-major matrix (pre-transposed
// weights / K^T / V^T): element (k, n) = M[n][k] -> contiguous 16-element load.
// lanes 0-15: N=n0+lane, K=k0+0..15 ; lanes 16-31: N=n0+lane-16, K=k0+16..31
__device__ __forceinline__ v16bf frag_bT(const __bf16* base, int ld, int k0, int n0, int lane) {
  const int kb = k0 + ((lane >> 4) & 1) * 16;
  const __bf16* p = base + (size_t)(n0 + (lane & 15)) * ld + kb;
  v16bf f;
#pragma unroll
  for (int i = 0; i < 16; ++i) f[i] = p[i];
  return f;
}

// C/D mapping: element r -> (M = m0 + 8*half + r, N = n0 + (lane & 15))

// Async global -> LDS 16-byte DMA (tracked with ASYNCcnt, no VGPR round-trip)
__device__ __forceinline__ void async_g2l_b128(const __bf16* g, __bf16* l) {
  unsigned lds = (unsigned)(uintptr_t)l;  // low 32 bits of flat LDS addr = LDS offset
  asm volatile("global_load_async_to_lds_b128 %0, %1, off"
               :: "v"(lds), "v"(g) : "memory");
}

// ---------------------------------------------------------------------------
// fp32 -> bf16 conversion with transpose: dst[N x K] from src[K x N]
// ---------------------------------------------------------------------------
__global__ void __launch_bounds__(256)
cvt_tr_f32_bf16(const float* __restrict__ s, __bf16* __restrict__ d, int K, int N) {
  int idx = blockIdx.x * blockDim.x + threadIdx.x;
  if (idx < K * N) {
    int k = idx / N, n = idx - k * N;
    d[(size_t)n * K + k] = (__bf16)s[idx];
  }
}

// ---------------------------------------------------------------------------
// LN (+ReLU) over `w` features for 16 rows, 128 threads (8 per row), bf16 out
// ---------------------------------------------------------------------------
__device__ __forceinline__ void ln_relu_store(const float* O, int w, const float* g,
                                              const float* be, float eps, __bf16* A,
                                              float* rs, float* rq, int tid) {
  const int row = tid >> 3;
  const int sub = tid & 7;
  const int chunk = w >> 3;
  const float* src = O + row * w + sub * chunk;
  float s = 0.f, q = 0.f;
  for (int i = 0; i < chunk; ++i) { float v = src[i]; s += v; q += v * v; }
  rs[row * 8 + sub] = s;
  rq[row * 8 + sub] = q;
  __syncthreads();
  float ts = 0.f, tq = 0.f;
#pragma unroll
  for (int i = 0; i < 8; ++i) { ts += rs[row * 8 + i]; tq += rq[row * 8 + i]; }
  const float mean = ts / (float)w;
  const float var  = tq / (float)w - mean * mean;
  const float inv  = rsqrtf(var + eps);
  __bf16* dst = A + row * w + sub * chunk;
  const float* gg = g + sub * chunk;
  const float* bb = be + sub * chunk;
  for (int i = 0; i < chunk; ++i) {
    float v = (src[i] - mean) * inv * gg[i] + bb[i];
    dst[i] = (__bf16)(v > 0.f ? v : 0.f);
  }
  __syncthreads();
}

// ---------------------------------------------------------------------------
// Fused retriever MLP: 16 tokens/block, 128 threads (4 waves), WMMA layers 1-3
// Weights are pre-transposed (N x K) so B fragments are contiguous 32B loads.
// ---------------------------------------------------------------------------
__global__ void __launch_bounds__(128)
retriever_kernel(const float* __restrict__ reprs, const float* __restrict__ times,
                 const __bf16* __restrict__ w1t, const float* __restrict__ w1f,
                 const float* __restrict__ b1, const float* __restrict__ g1, const float* __restrict__ be1,
                 const __bf16* __restrict__ w2t, const float* __restrict__ b2,
                 const float* __restrict__ g2, const float* __restrict__ be2,
                 const __bf16* __restrict__ w3t, const float* __restrict__ b3,
                 const float* __restrict__ g3, const float* __restrict__ be3,
                 const float* __restrict__ w4f, const float* __restrict__ b4,
                 float* __restrict__ sim, float* __restrict__ tadj) {
  const int tile = blockIdx.x;  // 0..519 (8320 / 16)
  const int b    = blockIdx.y;  // 0..31
  const int t0   = tile * 16;
  const int tid  = threadIdx.x;
  const int lane = tid & 31;
  const int wid  = tid >> 5;    // 0..3

  __shared__ __align__(16) __bf16 Ab[16 * 512];
  __shared__ float  Ob[16 * 256];
  __shared__ float  xt[16];
  __shared__ int    zf[16];     // 1 = row has a nonzero element
  __shared__ float  rs[16 * 8];
  __shared__ float  rq[16 * 8];

  if (tid < 16) zf[tid] = 0;
  __syncthreads();

  // ---- load 16x512 reprs (tiles are uniformly real (t0<8192) or padded) ----
  if (t0 < 8192) {
    const float* src = reprs + ((size_t)b * 8192 + t0) * 512;
    int curRow = 0, acc = 0;
    for (int idx = tid; idx < 16 * 512; idx += 128) {
      int row = idx >> 9;
      if (row != curRow) {
        if (acc) atomicOr(&zf[curRow], 1);
        acc = 0; curRow = row;
      }
      float v = src[idx];
      acc |= (v != 0.0f);
      Ab[idx] = (__bf16)v;
    }
    if (acc) atomicOr(&zf[curRow], 1);
  } else {
    for (int idx = tid; idx < 16 * 512; idx += 128) Ab[idx] = (__bf16)0.0f;
  }
  __syncthreads();
  if (tid < 16) {
    float tm = (t0 < 8192) ? times[(size_t)b * 8192 + t0 + tid] : 0.0f;
    float ta = zf[tid] ? tm : 1.0e10f;     // zero row -> 1e10
    xt[tid] = ta;
    tadj[(size_t)b * 8320 + t0 + tid] = ta;
  }
  __syncthreads();

  const int half = (lane >> 4) & 1;
  const int ncol = lane & 15;

  // ---- layer 1: [16x512]@[512x256] (+ time * w1[512,:]) ; LN ; ReLU ----
  {
    v8f acc[4];
    v8f z = {};
#pragma unroll
    for (int j = 0; j < 4; ++j) acc[j] = z;
    const int nbase = wid * 64;
    for (int kt = 0; kt < 512; kt += 32) {
      v16bf a = frag_a(Ab, 512, 0, kt, lane);
#pragma unroll
      for (int j = 0; j < 4; ++j) {
        v16bf bf_ = frag_bT(w1t, 512, kt, nbase + j * 16, lane);
        acc[j] = WMMA_BF16(a, bf_, acc[j]);
      }
    }
#pragma unroll
    for (int j = 0; j < 4; ++j) {
      int n = nbase + j * 16 + ncol;
      float wlast = w1f[512 * 256 + n];   // time row of w1 (513,256), exact fp32
      float bias  = b1[n];
#pragma unroll
      for (int r = 0; r < 8; ++r) {
        int m = half * 8 + r;
        Ob[m * 256 + n] = acc[j][r] + bias + xt[m] * wlast;
      }
    }
  }
  __syncthreads();
  ln_relu_store(Ob, 256, g1, be1, 1e-5f, Ab, rs, rq, tid);

  // ---- layer 2: [16x256]@[256x128] ; LN ; ReLU ----
  {
    v8f acc[2];
    v8f z = {};
    acc[0] = z; acc[1] = z;
    const int nbase = wid * 32;
    for (int kt = 0; kt < 256; kt += 32) {
      v16bf a = frag_a(Ab, 256, 0, kt, lane);
#pragma unroll
      for (int j = 0; j < 2; ++j) {
        v16bf bf_ = frag_bT(w2t, 256, kt, nbase + j * 16, lane);
        acc[j] = WMMA_BF16(a, bf_, acc[j]);
      }
    }
    __syncthreads();  // done reading Ab before LN rewrites it
#pragma unroll
    for (int j = 0; j < 2; ++j) {
      int n = nbase + j * 16 + ncol;
      float bias = b2[n];
#pragma unroll
      for (int r = 0; r < 8; ++r) Ob[(half * 8 + r) * 128 + n] = acc[j][r] + bias;
    }
  }
  __syncthreads();
  ln_relu_store(Ob, 128, g2, be2, 1e-5f, Ab, rs, rq, tid);

  // ---- layer 3: [16x128]@[128x64] ; LN ; ReLU ----
  {
    v8f acc = {};
    const int nbase = wid * 16;
    for (int kt = 0; kt < 128; kt += 32) {
      v16bf a = frag_a(Ab, 128, 0, kt, lane);
      v16bf bf_ = frag_bT(w3t, 128, kt, nbase, lane);
      acc = WMMA_BF16(a, bf_, acc);
    }
    __syncthreads();
    int n = nbase + ncol;
    float bias = b3[n];
#pragma unroll
    for (int r = 0; r < 8; ++r) Ob[(half * 8 + r) * 64 + n] = acc[r] + bias;
  }
  __syncthreads();
  ln_relu_store(Ob, 64, g3, be3, 1e-5f, Ab, rs, rq, tid);

  // ---- layer 4: dot(64) + b4 -> sigmoid ; mask zero rows ----
  if (tid < 16) {
    float s = b4[0];
    for (int i = 0; i < 64; ++i) s += (float)Ab[tid * 64 + i] * w4f[i];
    float sg = 1.0f / (1.0f + expf(-s));
    sim[(size_t)b * 8320 + t0 + tid] = zf[tid] ? sg : 0.0f;
  }
}

// ---------------------------------------------------------------------------
// Top-k (128 of 8320) per batch + stable sort by adjusted time
// ---------------------------------------------------------------------------
__global__ void __launch_bounds__(256)
topk_kernel(float* __restrict__ sim, const float* __restrict__ tadj,
            int* __restrict__ tk_idx, float* __restrict__ tk_time) {
  const int b = blockIdx.x;
  const int tid = threadIdx.x;
  float* s = sim + (size_t)b * 8320;
  __shared__ float bv[256];
  __shared__ int   bi_[256];
  __shared__ float selt[128];
  __shared__ int   seli[128];
  __shared__ int   selp[128];

  for (int it = 0; it < 128; ++it) {
    float mv = -2.0f; int mi = 0x7fffffff;
    for (int i = tid; i < 8320; i += 256) {
      float v = s[i];
      if (v > mv || (v == mv && i < mi)) { mv = v; mi = i; }
    }
    bv[tid] = mv; bi_[tid] = mi;
    __syncthreads();
    for (int off = 128; off > 0; off >>= 1) {
      if (tid < off) {
        float v2 = bv[tid + off]; int i2 = bi_[tid + off];
        if (v2 > bv[tid] || (v2 == bv[tid] && i2 < bi_[tid])) { bv[tid] = v2; bi_[tid] = i2; }
      }
      __syncthreads();
    }
    if (tid == 0) {
      int idx = bi_[0];
      seli[it] = idx; selp[it] = it;
      selt[it] = tadj[(size_t)b * 8320 + idx];
      s[idx] = -1.0f;
    }
    __syncthreads();
  }

  // bitonic sort of 128 by (time asc, original pos asc) — emulates stable argsort
  for (int ksz = 2; ksz <= 128; ksz <<= 1) {
    for (int j = ksz >> 1; j > 0; j >>= 1) {
      if (tid < 128) {
        int i = tid, p = i ^ j;
        if (p > i) {
          bool up = ((i & ksz) == 0);
          float t1 = selt[i], t2 = selt[p];
          int p1 = selp[i], p2 = selp[p];
          bool gt = (t1 > t2) || (t1 == t2 && p1 > p2);
          if (gt == up) {
            selt[i] = t2; selt[p] = t1;
            int ii = seli[i]; seli[i] = seli[p]; seli[p] = ii;
            selp[i] = p2; selp[p] = p1;
          }
        }
      }
      __syncthreads();
    }
  }
  if (tid < 128) {
    tk_idx[b * 128 + tid]  = seli[tid];
    tk_time[b * 128 + tid] = selt[tid];
  }
}

// ---------------------------------------------------------------------------
// Build x = topk (+rand_emb subst for row 0) + positional encoding; pad mask
// ---------------------------------------------------------------------------
__global__ void __launch_bounds__(256)
build_x_kernel(const float* __restrict__ reprs, const int* __restrict__ tk_idx,
               const float* __restrict__ tk_time, const float* __restrict__ rand_emb,
               float* __restrict__ xf, __bf16* __restrict__ xb, float* __restrict__ maskadd) {
  const int k = blockIdx.x;   // 0..127 (time-sorted slot)
  const int b = blockIdx.y;   // 0..31
  const int tid = threadIdx.x;
  __shared__ float red[256];
  __shared__ int   anyr;

  int idx = tk_idx[b * 128 + k];
  float v0 = 0.f, v1 = 0.f;
  if (idx < 8192) {
    const float* src = reprs + ((size_t)b * 8192 + idx) * 512;
    v0 = src[tid]; v1 = src[tid + 256];
  }
  red[tid] = v0 + v1;
  if (tid == 0) anyr = 0;
  __syncthreads();
  if ((v0 != 0.f) || (v1 != 0.f)) atomicOr(&anyr, 1);
  for (int off = 128; off > 0; off >>= 1) {
    if (tid < off) red[tid] += red[tid + off];
    __syncthreads();
  }
  float rowsum = red[0];
  bool subst = (k == 0) && (rowsum == 0.0f);
  float w0 = subst ? rand_emb[tid] : v0;
  float w1 = subst ? rand_emb[tid + 256] : v1;
  bool anyAfter = subst ? true : (anyr != 0);
  if (tid == 0) maskadd[b * 128 + k] = anyAfter ? 0.0f : -3.402823466e38f;

  float t = 48.0f * 60.0f - tk_time[b * 128 + k];
  auto pe = [&](int j) -> float {
    int i = j >> 1;
    float div = expf((float)(2 * i) * (-9.210340371976184f / 512.0f));
    float ang = t * div;
    return (j & 1) ? cosf(ang) : sinf(ang);
  };
  size_t base = ((size_t)b * 128 + k) * 512;
  float o0 = w0 + pe(tid), o1 = w1 + pe(tid + 256);
  xf[base + tid] = o0;        xf[base + tid + 256] = o1;
  xb[base + tid] = (__bf16)o0; xb[base + tid + 256] = (__bf16)o1;
}

// ---------------------------------------------------------------------------
// Generic bf16 WMMA GEMM, 64x64 tiles, double-buffered async global->LDS DMA.
// A: M x K row-major; Bt: N x K row-major (pre-transposed); C = act(A@B + bias)
// cbT==1: bf16 store transposed per 128-row batch: Cb[(b*N + n)*128 + row]
// ---------------------------------------------------------------------------
__global__ void __launch_bounds__(256)
gemm_bf16_kernel(const __bf16* __restrict__ A, const __bf16* __restrict__ Bt,
                 const float* __restrict__ bias, float* __restrict__ Cf,
                 __bf16* __restrict__ Cb, int M, int N, int K, int act, int cbT) {
  __shared__ __align__(16) __bf16 As[2][64 * 32];
  __shared__ __align__(16) __bf16 Bs[2][64 * 32];   // n-major: [n][k]
  const int tid = threadIdx.x, lane = tid & 31, wid = tid >> 5;
  const int bm = blockIdx.x * 64, bn = blockIdx.y * 64;
  const int m0 = (wid & 3) * 16, n0 = (wid >> 2) * 32;
  v8f acc0 = {}, acc1 = {};

  // staging: 64 rows x 32 cols, 8 contiguous bf16 (16B = one async b128) / thread
  const int r8 = tid >> 2, c8 = (tid & 3) * 8;
  const __bf16* gA = A  + (size_t)(bm + r8) * K + c8;
  const __bf16* gB = Bt + (size_t)(bn + r8) * K + c8;
  const int lo = r8 * 32 + c8;

  async_g2l_b128(gA, &As[0][lo]);
  async_g2l_b128(gB, &Bs[0][lo]);
  int buf = 0;
  for (int kt = 0; kt < K; kt += 32) {
    const int nxt = buf ^ 1;
    if (kt + 32 < K) {
      async_g2l_b128(gA + kt + 32, &As[nxt][lo]);
      async_g2l_b128(gB + kt + 32, &Bs[nxt][lo]);
      asm volatile("s_wait_asynccnt 2" ::: "memory");  // current tile landed
    } else {
      asm volatile("s_wait_asynccnt 0" ::: "memory");
    }
    __syncthreads();
    v16bf a  = frag_a (As[buf], 32, m0, 0, lane);
    v16bf b0 = frag_bT(Bs[buf], 32, 0, n0, lane);
    v16bf b1 = frag_bT(Bs[buf], 32, 0, n0 + 16, lane);
    acc0 = WMMA_BF16(a, b0, acc0);
    acc1 = WMMA_BF16(a, b1, acc1);
    __syncthreads();
    buf = nxt;
  }

  const int half = (lane >> 4) & 1, nc = lane & 15;
#pragma unroll
  for (int j = 0; j < 2; ++j) {
    const v8f& acc = j ? acc1 : acc0;
    int n = bn + n0 + j * 16 + nc;
    float bs = bias ? bias[n] : 0.0f;
#pragma unroll
    for (int r = 0; r < 8; ++r) {
      int m = bm + m0 + half * 8 + r;
      float v = acc[r] + bs;
      if (act == 1) v = 0.5f * v * (1.0f + erff(v * 0.7071067811865476f));  // exact GELU
      if (Cf) Cf[(size_t)m * N + n] = v;
      if (Cb) {
        size_t ci = cbT ? ((size_t)((m >> 7) * N + n) * 128 + (m & 127))
                        : ((size_t)m * N + n);
        Cb[ci] = (__bf16)v;
      }
    }
  }
}

// ---------------------------------------------------------------------------
// Rotary embedding on q,k (interleaved pairs); 1/sqrt(DH) folded into q
// ---------------------------------------------------------------------------
__global__ void __launch_bounds__(256)
rotary_kernel(const float* __restrict__ qf, const float* __restrict__ kf,
              __bf16* __restrict__ qb, __bf16* __restrict__ kb) {
  const int row = blockIdx.x;      // b*128 + pos
  const int pos = row & 127;
  const int tid = threadIdx.x;     // 8 heads x 32 pairs
  const int p = tid & 31, h = tid >> 5;
  const int j = h * 64 + 2 * p;
  const size_t base = (size_t)row * 512;
  float fr  = expf((float)(2 * p) * (-9.210340371976184f / 64.0f));
  float ang = (float)pos * fr;
  float cs = cosf(ang), sn = sinf(ang);
  float q0 = qf[base + j], q1 = qf[base + j + 1];
  float k0 = kf[base + j], k1 = kf[base + j + 1];
  const float scale = 0.125f;  // 1/sqrt(64)
  qb[base + j]     = (__bf16)((q0 * cs - q1 * sn) * scale);
  qb[base + j + 1] = (__bf16)((q1 * cs + q0 * sn) * scale);
  kb[base + j]     = (__bf16)(k0 * cs - k1 * sn);
  kb[base + j + 1] = (__bf16)(k1 * cs + k0 * sn);
}

// ---------------------------------------------------------------------------
// Attention per (b,h): scores = q@k^T + mask(query rows); softmax; ctx = P@v
// vt is batch-transposed: vt[(b*512 + d) * 128 + token]  -> contiguous B-frags
// ---------------------------------------------------------------------------
__global__ void __launch_bounds__(256)
attention_kernel(const __bf16* __restrict__ qb, const __bf16* __restrict__ kb,
                 const __bf16* __restrict__ vt, const float* __restrict__ maskadd,
                 __bf16* __restrict__ ctxb) {
  const int h = blockIdx.x & 7;
  const int b = blockIdx.x >> 3;
  const int tid = threadIdx.x, lane = tid & 31, wid = tid >> 5;
  __shared__ float sc[128 * 128];  // 64 KB; P (bf16) packed in-place per row
  const __bf16* Q  = qb + (size_t)b * 128 * 512 + h * 64;
  const __bf16* Km = kb + (size_t)b * 128 * 512 + h * 64;
  const __bf16* Vt = vt + ((size_t)b * 512 + h * 64) * 128;
  const int half = (lane >> 4) & 1, nc = lane & 15;

  // ---- scores: wave wid handles rows m0=wid*16, all 8 column tiles ----
  {
    const int m0 = wid * 16;
    v8f acc[8];
    v8f z = {};
#pragma unroll
    for (int j = 0; j < 8; ++j) acc[j] = z;
    for (int kt = 0; kt < 64; kt += 32) {
      v16bf a = frag_a(Q, 512, m0, kt, lane);
#pragma unroll
      for (int j = 0; j < 8; ++j) {
        v16bf bt = frag_bT(Km, 512, kt, j * 16, lane);  // K^T: contiguous loads
        acc[j] = WMMA_BF16(a, bt, acc[j]);
      }
    }
#pragma unroll
    for (int j = 0; j < 8; ++j) {
      int n = j * 16 + nc;
#pragma unroll
      for (int r = 0; r < 8; ++r) {
        int m = m0 + half * 8 + r;
        sc[m * 128 + n] = acc[j][r] + maskadd[b * 128 + m];  // mask per query row
      }
    }
  }
  __syncthreads();

  // ---- softmax per row, pack P as bf16 into row start (in-place, safe order)
  if (tid < 128) {
    float* row = sc + tid * 128;
    float mx = row[0];
    for (int i = 1; i < 128; ++i) mx = fmaxf(mx, row[i]);
    float sum = 0.0f;
    for (int i = 0; i < 128; ++i) sum += expf(row[i] - mx);
    float inv = 1.0f / sum;
    __bf16* prow = (__bf16*)row;
    for (int i = 0; i < 128; ++i) {
      float pv = expf(row[i] - mx) * inv;
      prow[i] = (__bf16)pv;
    }
  }
  __syncthreads();

  // ---- ctx = P(128x128) @ V(128x64); wave wid: rows wid*16, 4 n-tiles ----
  {
    const int m0 = wid * 16;
    v8f acc[4];
    v8f z = {};
#pragma unroll
    for (int j = 0; j < 4; ++j) acc[j] = z;
    const __bf16* P = (const __bf16*)sc;  // ld = 256 bf16 slots per row
    for (int kt = 0; kt < 128; kt += 32) {
      v16bf a = frag_a(P, 256, m0, kt, lane);
#pragma unroll
      for (int j = 0; j < 4; ++j) {
        v16bf bf_ = frag_bT(Vt, 128, kt, j * 16, lane);  // V^T: contiguous loads
        acc[j] = WMMA_BF16(a, bf_, acc[j]);
      }
    }
    __bf16* O = ctxb + (size_t)b * 128 * 512 + h * 64;
#pragma unroll
    for (int j = 0; j < 4; ++j) {
      int n = j * 16 + nc;
#pragma unroll
      for (int r = 0; r < 8; ++r) {
        int m = m0 + half * 8 + r;
        O[(size_t)m * 512 + n] = (__bf16)acc[j][r];
      }
    }
  }
}

// ---------------------------------------------------------------------------
// x = LN(x + dx) * g + be (eps=1e-12), write f32 + bf16
// ---------------------------------------------------------------------------
__global__ void __launch_bounds__(256)
resid_ln_kernel(float* __restrict__ xf, __bf16* __restrict__ xb,
                const float* __restrict__ dx, const float* __restrict__ g,
                const float* __restrict__ be) {
  const int row = blockIdx.x;  // 0..4095
  const int tid = threadIdx.x;
  __shared__ float red[256], red2[256];
  const size_t base = (size_t)row * 512;
  float v0 = xf[base + tid] + dx[base + tid];
  float v1 = xf[base + tid + 256] + dx[base + tid + 256];
  red[tid] = v0 + v1;
  red2[tid] = v0 * v0 + v1 * v1;
  __syncthreads();
  for (int off = 128; off > 0; off >>= 1) {
    if (tid < off) { red[tid] += red[tid + off]; red2[tid] += red2[tid + off]; }
    __syncthreads();
  }
  float mean = red[0] / 512.0f;
  float var  = red2[0] / 512.0f - mean * mean;
  float inv  = rsqrtf(var + 1e-12f);
  float o0 = (v0 - mean) * inv * g[tid] + be[tid];
  float o1 = (v1 - mean) * inv * g[tid + 256] + be[tid + 256];
  xf[base + tid] = o0;        xf[base + tid + 256] = o1;
  xb[base + tid] = (__bf16)o0; xb[base + tid + 256] = (__bf16)o1;
}

// ---------------------------------------------------------------------------
// Masked mean-pool over 128 tokens + final linear + sigmoid
// ---------------------------------------------------------------------------
__global__ void __launch_bounds__(256)
pool_final_kernel(const float* __restrict__ xf, const float* __restrict__ fw,
                  const float* __restrict__ fb, float* __restrict__ out) {
  const int b = blockIdx.x;
  const int tid = threadIdx.x;
  __shared__ int rowany[128];
  __shared__ float red[256];
  const float* X = xf + (size_t)b * 128 * 512;
  if (tid < 128) {
    int a = 0;
    const float* r = X + (size_t)tid * 512;
    for (int i = 0; i < 512; ++i) a |= (r[i] != 0.0f);
    rowany[tid] = a;
  }
  __syncthreads();
  float s0 = 0.f, s1 = 0.f; int cnt = 0;
  for (int k = 0; k < 128; ++k) {
    if (rowany[k]) {
      s0 += X[(size_t)k * 512 + tid];
      s1 += X[(size_t)k * 512 + tid + 256];
      cnt++;
    }
  }
  float fc = (float)cnt;
  red[tid] = (s0 / fc) * fw[tid] + (s1 / fc) * fw[tid + 256];
  __syncthreads();
  for (int off = 128; off > 0; off >>= 1) {
    if (tid < off) red[tid] += red[tid + off];
    __syncthreads();
  }
  if (tid == 0) out[b] = 1.0f / (1.0f + expf(-(red[0] + fb[0])));
}

// ---------------------------------------------------------------------------
// Host orchestration. Input mapping = JAX pytree leaf order (dict keys sorted
// at every level, lists in order):
//   0 final_b, 1 final_w, 2..17 layer0{be1,be2,bi,bk,bo,bo2,bq,bv,g1,g2,wi,wk,
//   wo,wo2,wq,wv}, 18..33 layer1{...}, 34 rand_emb, 35..48 retr{b1,b2,b3,b4,
//   be1,be2,be3,g1,g2,g3,w1,w2,w3,w4}, 49 reprs, 50 times.
// Workspace requirement: ~87 MB.
// ---------------------------------------------------------------------------
extern "C" void kernel_launch(void* const* d_in, const int* in_sizes, int n_in,
                              void* d_out, int out_size, void* d_ws, size_t ws_size,
                              hipStream_t stream) {
  (void)in_sizes; (void)n_in; (void)out_size; (void)ws_size;

  const float* final_b = (const float*)d_in[0];
  const float* final_w = (const float*)d_in[1];
  const float *Lbe1[2], *Lbe2[2], *Lbi[2], *Lbk[2], *Lbo[2], *Lbo2[2], *Lbq[2], *Lbv[2];
  const float *Lg1[2], *Lg2[2], *Lwi[2], *Lwk[2], *Lwo[2], *Lwo2[2], *Lwq[2], *Lwv[2];
  for (int l = 0; l < 2; ++l) {
    const int o = 2 + l * 16;
    Lbe1[l] = (const float*)d_in[o + 0];  Lbe2[l] = (const float*)d_in[o + 1];
    Lbi[l]  = (const float*)d_in[o + 2];  Lbk[l]  = (const float*)d_in[o + 3];
    Lbo[l]  = (const float*)d_in[o + 4];  Lbo2[l] = (const float*)d_in[o + 5];
    Lbq[l]  = (const float*)d_in[o + 6];  Lbv[l]  = (const float*)d_in[o + 7];
    Lg1[l]  = (const float*)d_in[o + 8];  Lg2[l]  = (const float*)d_in[o + 9];
    Lwi[l]  = (const float*)d_in[o + 10]; Lwk[l]  = (const float*)d_in[o + 11];
    Lwo[l]  = (const float*)d_in[o + 12]; Lwo2[l] = (const float*)d_in[o + 13];
    Lwq[l]  = (const float*)d_in[o + 14]; Lwv[l]  = (const float*)d_in[o + 15];
  }
  const float* rand_emb = (const float*)d_in[34];
  const float* r_b1  = (const float*)d_in[35];
  const float* r_b2  = (const float*)d_in[36];
  const float* r_b3  = (const float*)d_in[37];
  const float* r_b4  = (const float*)d_in[38];
  const float* r_be1 = (const float*)d_in[39];
  const float* r_be2 = (const float*)d_in[40];
  const float* r_be3 = (const float*)d_in[41];
  const float* r_g1  = (const float*)d_in[42];
  const float* r_g2  = (const float*)d_in[43];
  const float* r_g3  = (const float*)d_in[44];
  const float* r_w1  = (const float*)d_in[45];
  const float* r_w2  = (const float*)d_in[46];
  const float* r_w3  = (const float*)d_in[47];
  const float* r_w4  = (const float*)d_in[48];
  const float* reprs = (const float*)d_in[49];
  const float* times = (const float*)d_in[50];

  // ---- workspace carve ----
  char* wp = (char*)d_ws;
  auto take = [&](size_t bytes) -> void* {
    void* r = (void*)wp;
    wp += (bytes + 255) & ~(size_t)255;
    return r;
  };
  __bf16* w1t = (__bf16*)take((size_t)256 * 512 * 2);   // transposed, time row excl.
  __bf16* w2t = (__bf16*)take((size_t)128 * 256 * 2);
  __bf16* w3t = (__bf16*)take((size_t)64 * 128 * 2);
  __bf16 *wqT[2], *wkT[2], *wvT[2], *woT[2], *wiT[2], *wo2T[2];
  for (int l = 0; l < 2; ++l) {
    wqT[l]  = (__bf16*)take((size_t)512 * 512 * 2);
    wkT[l]  = (__bf16*)take((size_t)512 * 512 * 2);
    wvT[l]  = (__bf16*)take((size_t)512 * 512 * 2);
    woT[l]  = (__bf16*)take((size_t)512 * 512 * 2);
    wiT[l]  = (__bf16*)take((size_t)2048 * 512 * 2);    // (N=2048) x (K=512)
    wo2T[l] = (__bf16*)take((size_t)512 * 2048 * 2);    // (N=512) x (K=2048)
  }
  float*  sim     = (float*)take((size_t)32 * 8320 * 4);
  float*  tadj    = (float*)take((size_t)32 * 8320 * 4);
  int*    tk_idx  = (int*)take((size_t)32 * 128 * 4);
  float*  tk_time = (float*)take((size_t)32 * 128 * 4);
  float*  maskadd = (float*)take((size_t)32 * 128 * 4);
  float*  x_f     = (float*)take((size_t)32 * 128 * 512 * 4);
  float*  qf      = (float*)take((size_t)32 * 128 * 512 * 4);
  float*  kf      = (float*)take((size_t)32 * 128 * 512 * 4);
  float*  tmp     = (float*)take((size_t)32 * 128 * 512 * 4);
  __bf16* x_b     = (__bf16*)take((size_t)32 * 128 * 512 * 2);
  __bf16* q_b     = (__bf16*)take((size_t)32 * 128 * 512 * 2);
  __bf16* k_b     = (__bf16*)take((size_t)32 * 128 * 512 * 2);
  __bf16* v_t     = (__bf16*)take((size_t)32 * 512 * 128 * 2);  // batch-transposed V
  __bf16* ctx_b   = (__bf16*)take((size_t)32 * 128 * 512 * 2);
  __bf16* h_b     = (__bf16*)take((size_t)32 * 128 * 2048 * 2);

  auto cvt_tr = [&](const float* s, __bf16* d, int K, int N) {
    cvt_tr_f32_bf16<<<(K * N + 255) / 256, 256, 0, stream>>>(s, d, K, N);
  };
  cvt_tr(r_w1, w1t, 512, 256);   // rows 0..511 only; time row applied in fp32
  cvt_tr(r_w2, w2t, 256, 128);
  cvt_tr(r_w3, w3t, 128, 64);
  for (int l = 0; l < 2; ++l) {
    cvt_tr(Lwq[l],  wqT[l],  512, 512);
    cvt_tr(Lwk[l],  wkT[l],  512, 512);
    cvt_tr(Lwv[l],  wvT[l],  512, 512);
    cvt_tr(Lwo[l],  woT[l],  512, 512);
    cvt_tr(Lwi[l],  wiT[l],  512, 2048);
    cvt_tr(Lwo2[l], wo2T[l], 2048, 512);
  }

  // ---- retriever: sim + adjusted times ----
  retriever_kernel<<<dim3(520, 32), 128, 0, stream>>>(
      reprs, times, w1t, r_w1, r_b1, r_g1, r_be1, w2t, r_b2, r_g2, r_be2,
      w3t, r_b3, r_g3, r_be3, r_w4, r_b4, sim, tadj);

  // ---- top-k + time sort ----
  topk_kernel<<<32, 256, 0, stream>>>(sim, tadj, tk_idx, tk_time);

  // ---- build x (+PE, rand_emb substitution, pad mask) ----
  build_x_kernel<<<dim3(128, 32), 256, 0, stream>>>(reprs, tk_idx, tk_time, rand_emb,
                                                    x_f, x_b, maskadd);

  auto gemm = [&](const __bf16* A, const __bf16* Bt, const float* bias, float* Cf,
                  __bf16* Cb, int M, int N, int K, int act, int cbT) {
    gemm_bf16_kernel<<<dim3(M / 64, N / 64), 256, 0, stream>>>(A, Bt, bias, Cf, Cb,
                                                               M, N, K, act, cbT);
  };

  const int M = 32 * 128;  // 4096 token rows, batch-flattened
  for (int l = 0; l < 2; ++l) {
    gemm(x_b, wqT[l], Lbq[l], qf, nullptr, M, 512, 512, 0, 0);
    gemm(x_b, wkT[l], Lbk[l], kf, nullptr, M, 512, 512, 0, 0);
    gemm(x_b, wvT[l], Lbv[l], nullptr, v_t, M, 512, 512, 0, 1);  // transposed bf16 V
    rotary_kernel<<<M, 256, 0, stream>>>(qf, kf, q_b, k_b);
    attention_kernel<<<32 * 8, 256, 0, stream>>>(q_b, k_b, v_t, maskadd, ctx_b);
    gemm(ctx_b, woT[l], Lbo[l], tmp, nullptr, M, 512, 512, 0, 0);
    resid_ln_kernel<<<M, 256, 0, stream>>>(x_f, x_b, tmp, Lg1[l], Lbe1[l]);
    gemm(x_b, wiT[l], Lbi[l], nullptr, h_b, M, 2048, 512, 1 /*gelu*/, 0);
    gemm(h_b, wo2T[l], Lbo2[l], tmp, nullptr, M, 512, 2048, 0, 0);
    resid_ln_kernel<<<M, 256, 0, stream>>>(x_f, x_b, tmp, Lg2[l], Lbe2[l]);
  }

  pool_final_kernel<<<32, 256, 0, stream>>>(x_f, final_w, final_b, (float*)d_out);
}
